// VecdiffDecoder_26972394619470
// MI455X (gfx1250) — matrix operand order repaired and checked by reading, not compiled, not run on hardware
//
#include <hip/hip_runtime.h>
#include <hip/hip_bf16.h>

typedef __bf16 bf16_t;
typedef __attribute__((ext_vector_type(16))) __bf16 v16bf;
typedef __attribute__((ext_vector_type(8)))  __bf16 v8bf;
typedef __attribute__((ext_vector_type(8)))  float  v8f;
typedef __attribute__((ext_vector_type(4)))  int    v4i;

typedef __attribute__((address_space(1))) void as1_void;
typedef __attribute__((address_space(3))) void as3_void;
typedef __attribute__((address_space(1))) v4i  as1_v4i;
typedef __attribute__((address_space(3))) v4i  as3_v4i;

#define Bn   4096
#define Sn   64
#define Dn   512
#define K1   1024   // 2*D, GEMM K for fg/ig and og
#define KSTG 64     // K per LDS stage
#define NSTG (K1 / KSTG)   // 16 stages

#if __has_builtin(__builtin_amdgcn_global_load_async_to_lds_b128)
#define ASYNC_COPY 1
#endif

__device__ __forceinline__ float sigmoidf_(float x) {
    return 1.0f / (1.0f + __expf(-x));
}

// ---- async (memory -> LDS) copy of one 16-byte chunk per lane --------------
__device__ __forceinline__ void cp_b128(const bf16_t* g, bf16_t* l) {
#ifdef ASYNC_COPY
    __builtin_amdgcn_global_load_async_to_lds_b128(
        (as1_v4i*)(as1_void*)(void*)(const void*)g,
        (as3_v4i*)(as3_void*)(void*)l, 0, 0);
#else
    *(uint4*)l = *(const uint4*)g;   // global_load_b128 + ds_store_b128
#endif
}

template <int N>
__device__ __forceinline__ void wait_async() {
#ifdef ASYNC_COPY
#if __has_builtin(__builtin_amdgcn_s_wait_asynccnt)
    __builtin_amdgcn_s_wait_asynccnt(N);
#else
    asm volatile("s_wait_asynccnt %0" :: "i"(N) : "memory");
#endif
#endif
}

// ---------------------------------------------------------------------------
// Kernel 1: per-row cosine similarities + bf16 conversion of activations.
// One wave per row b.
// ---------------------------------------------------------------------------
__global__ void __launch_bounds__(256) prep_rows(
    const int* __restrict__ ip,
    const float* __restrict__ src,         // (B, S, D)
    const float* __restrict__ enc_memory,  // (S, B, 2D)
    const float* __restrict__ A,           // (B, D)
    const float* __restrict__ sv,          // (B, D)
    bf16_t* __restrict__ enc16,            // (B, 1024)
    bf16_t* __restrict__ BA16,             // (B, 1024) = [Bv | A]
    float* __restrict__ prev_cur,
    float* __restrict__ prev_tot)
{
    const int i    = ip[0];
    const int wave = threadIdx.x >> 5;
    const int lane = threadIdx.x & 31;
    const int b    = blockIdx.x * 8 + wave;

    const float* Ar  = A  + (size_t)b * Dn;
    const float* Bvr = src + ((size_t)b * Sn + i) * Dn;
    const float* svr = sv + (size_t)b * Dn;

    float dAB = 0.f, dAS = 0.f, sA = 0.f, sB = 0.f, sS = 0.f;
    const int e0 = lane * 16;
#pragma unroll
    for (int e = 0; e < 16; ++e) {
        float a  = Ar[e0 + e];
        float bv = Bvr[e0 + e];
        float s  = svr[e0 + e];
        dAB += a * bv; dAS += a * s;
        sA  += a * a;  sB  += bv * bv; sS += s * s;
        BA16[(size_t)b * K1 + e0 + e]      = (bf16_t)bv;
        BA16[(size_t)b * K1 + Dn + e0 + e] = (bf16_t)a;
    }

    const float* er = enc_memory + ((size_t)i * Bn + b) * K1;
    const int e1 = lane * 32;
#pragma unroll
    for (int e = 0; e < 32; ++e)
        enc16[(size_t)b * K1 + e1 + e] = (bf16_t)er[e1 + e];

#pragma unroll
    for (int off = 16; off >= 1; off >>= 1) {
        dAB += __shfl_xor(dAB, off, 32);
        dAS += __shfl_xor(dAS, off, 32);
        sA  += __shfl_xor(sA,  off, 32);
        sB  += __shfl_xor(sB,  off, 32);
        sS  += __shfl_xor(sS,  off, 32);
    }
    if (lane == 0) {
        float na = fmaxf(sqrtf(sA), 1e-8f);
        float nb = fmaxf(sqrtf(sB), 1e-8f);
        float ns = fmaxf(sqrtf(sS), 1e-8f);
        prev_cur[b] = dAB / (na * nb);
        prev_tot[b] = dAS / (na * ns);
    }
}

// ---------------------------------------------------------------------------
// Kernel 2: weights -> bf16. Wog16 = [Wogb | WogA] fused over K=1024.
// ---------------------------------------------------------------------------
__global__ void __launch_bounds__(256) conv_weights(
    const float* __restrict__ Wfg_w,   // (512, 1026)
    const float* __restrict__ Wig_w,   // (512, 1026)
    const float* __restrict__ Wogb_w,  // (512, 512)
    const float* __restrict__ WogA_w,  // (512, 512)
    const float* __restrict__ Wctb_w,  // (512, 512)
    bf16_t* __restrict__ Wfg16,
    bf16_t* __restrict__ Wig16,
    bf16_t* __restrict__ Wog16,
    bf16_t* __restrict__ Wctb16)
{
    const int idx = blockIdx.x * blockDim.x + threadIdx.x;
    const int n = idx >> 10;
    const int k = idx & 1023;
    Wfg16[idx] = (bf16_t)Wfg_w[(size_t)n * 1026 + k];
    Wig16[idx] = (bf16_t)Wig_w[(size_t)n * 1026 + k];
    Wog16[idx] = (bf16_t)(k < Dn ? Wogb_w[(size_t)n * Dn + k]
                                 : WogA_w[(size_t)n * Dn + k - Dn]);
    if (k < Dn)
        Wctb16[(size_t)n * Dn + k] = (bf16_t)Wctb_w[(size_t)n * Dn + k];
}

// ---------------------------------------------------------------------------
// WMMA fragment loaders (row-major bf16 source with row stride ldk).
// A 16x32: lane l -> row, 16 contiguous K at k0 + (hi?16:0).
// B 32x16: lane l -> col n, 16 contiguous K at k0 + (hi?16:0).
// ---------------------------------------------------------------------------
__device__ __forceinline__ v16bf a_frag(const bf16_t* __restrict__ rowp,
                                        int k0, bool hi)
{
    const bf16_t* p = rowp + k0 + (hi ? 16 : 0);
    v8bf c0 = *(const v8bf*)(p);
    v8bf c1 = *(const v8bf*)(p + 8);
    v16bf a;
#pragma unroll
    for (int e = 0; e < 8; ++e) { a[e] = c0[e]; a[e + 8] = c1[e]; }
    return a;
}

__device__ __forceinline__ v16bf b_frag(const bf16_t* __restrict__ W,
                                        int n, int ldk, int k0, bool hi)
{
    return *(const v16bf*)(W + (size_t)n * ldk + k0 + (hi ? 16 : 0));
}

#define WMMA_BF16(acc, a, b) \
    __builtin_amdgcn_wmma_f32_16x16x32_bf16(false, (a), false, (b), (short)0, (acc), false, false)

// ---------------------------------------------------------------------------
// Kernel 3: fg/ig GEMMs with LDS double-buffered staging (async -> LDS).
// Block = 128 threads (4 waves), 64 rows x 64 cols, both outputs.
// Per stage: A 64x64, Wfg 64x64, Wig 64x64 (8KB each), 12 b128 copies/thread.
// ---------------------------------------------------------------------------
__global__ void __launch_bounds__(128) gemm_fgig(
    const bf16_t* __restrict__ enc16,
    const bf16_t* __restrict__ Wfg16, const bf16_t* __restrict__ Wig16,
    const float* __restrict__ Wfg_w,  const float* __restrict__ Wig_w,
    const float* __restrict__ Wfg_b,  const float* __restrict__ Wig_b,
    const float* __restrict__ prev_cur, const float* __restrict__ prev_tot,
    float* __restrict__ fg, float* __restrict__ ig)
{
    __shared__ __align__(16) bf16_t shA[2][64 * KSTG];
    __shared__ __align__(16) bf16_t shF[2][64 * KSTG];
    __shared__ __align__(16) bf16_t shI[2][64 * KSTG];

    const int tid  = threadIdx.x;
    const int lane = tid & 31;
    const int wave = tid >> 5;
    const int m0   = blockIdx.x * 64;
    const int n0   = blockIdx.y * 64;
    const int l16  = lane & 15;
    const bool hi  = lane >= 16;

    v8f accF[4] = {{0}}; v8f accI[4] = {{0}};

    auto stage_in = [&](int buf, int s) {
        const int k0 = s * KSTG;
#pragma unroll
        for (int j = 0; j < 4; ++j) {
            const int t  = tid + j * 128;        // 0..511
            const int r  = t >> 3;
            const int c8 = (t & 7) * 8;
            cp_b128(enc16 + (size_t)(m0 + r) * K1 + k0 + c8, &shA[buf][r * KSTG + c8]);
            cp_b128(Wfg16 + (size_t)(n0 + r) * K1 + k0 + c8, &shF[buf][r * KSTG + c8]);
            cp_b128(Wig16 + (size_t)(n0 + r) * K1 + k0 + c8, &shI[buf][r * KSTG + c8]);
        }
    };

    stage_in(0, 0);
    for (int s = 0; s < NSTG; ++s) {
        const int buf = s & 1;
        if (s + 1 < NSTG) { stage_in(buf ^ 1, s + 1); wait_async<12>(); }
        else              { wait_async<0>(); }
        __syncthreads();

        const bf16_t* arow = &shA[buf][(wave * 16 + l16) * KSTG];
#pragma unroll
        for (int ks = 0; ks < 2; ++ks) {
            const int kk = ks * 32;
            v16bf a = a_frag(arow, kk, hi);
#pragma unroll
            for (int t = 0; t < 4; ++t) {
                const int nt = t * 16 + l16;
                v16bf bF = b_frag(&shF[buf][0], nt, KSTG, kk, hi);
                v16bf bI = b_frag(&shI[buf][0], nt, KSTG, kk, hi);
                accF[t] = WMMA_BF16(accF[t], a, bF);
                accI[t] = WMMA_BF16(accI[t], a, bI);
            }
        }
        __syncthreads();
    }

    // epilogue: bias + rank-2 cosine correction + sigmoid
    float pc[8], pt[8];
#pragma unroll
    for (int r = 0; r < 8; ++r) {
        const int m = m0 + wave * 16 + r + (hi ? 8 : 0);
        pc[r] = prev_cur[m]; pt[r] = prev_tot[m];
    }
#pragma unroll
    for (int t = 0; t < 4; ++t) {
        const int n = n0 + t * 16 + l16;
        const float wcF = Wfg_w[(size_t)n * 1026 + 1024];
        const float wtF = Wfg_w[(size_t)n * 1026 + 1025];
        const float bF  = Wfg_b[n];
        const float wcI = Wig_w[(size_t)n * 1026 + 1024];
        const float wtI = Wig_w[(size_t)n * 1026 + 1025];
        const float bI  = Wig_b[n];
#pragma unroll
        for (int r = 0; r < 8; ++r) {
            const int m = m0 + wave * 16 + r + (hi ? 8 : 0);
            float vF = accF[t][r] + bF + pc[r] * wcF + pt[r] * wtF;
            float vI = accI[t][r] + bI + pc[r] * wcI + pt[r] * wtI;
            fg[(size_t)m * Dn + n] = sigmoidf_(vF);
            ig[(size_t)m * Dn + n] = sigmoidf_(vI);
        }
    }
}

// ---------------------------------------------------------------------------
// Kernel 4: og = sigmoid([Bv|A] @ [Wogb|WogA]^T + b) (K=1024)
//           ctb = tanh(Bv @ Wctb^T + b)              (K=512, stages 0..7)
// A and Wog staged through LDS (8 copies/thread/stage); Wctb fragments come
// straight from global (0.5MB, L2-resident).
// ---------------------------------------------------------------------------
__global__ void __launch_bounds__(128) gemm_ogct(
    const bf16_t* __restrict__ BA16,
    const bf16_t* __restrict__ Wog16, const bf16_t* __restrict__ Wctb16,
    const float* __restrict__ Wogb_b, const float* __restrict__ WogA_b,
    const float* __restrict__ Wctb_b,
    float* __restrict__ og, float* __restrict__ ctb)
{
    __shared__ __align__(16) bf16_t shA[2][64 * KSTG];
    __shared__ __align__(16) bf16_t shO[2][64 * KSTG];

    const int tid  = threadIdx.x;
    const int lane = tid & 31;
    const int wave = tid >> 5;
    const int m0   = blockIdx.x * 64;
    const int n0   = blockIdx.y * 64;
    const int l16  = lane & 15;
    const bool hi  = lane >= 16;

    v8f accO[4] = {{0}}; v8f accC[4] = {{0}};

    auto stage_in = [&](int buf, int s) {
        const int k0 = s * KSTG;
#pragma unroll
        for (int j = 0; j < 4; ++j) {
            const int t  = tid + j * 128;
            const int r  = t >> 3;
            const int c8 = (t & 7) * 8;
            cp_b128(BA16  + (size_t)(m0 + r) * K1 + k0 + c8, &shA[buf][r * KSTG + c8]);
            cp_b128(Wog16 + (size_t)(n0 + r) * K1 + k0 + c8, &shO[buf][r * KSTG + c8]);
        }
    };

    stage_in(0, 0);
    for (int s = 0; s < NSTG; ++s) {
        const int buf = s & 1;
        if (s + 1 < NSTG) { stage_in(buf ^ 1, s + 1); wait_async<8>(); }
        else              { wait_async<0>(); }
        __syncthreads();

        const bf16_t* arow = &shA[buf][(wave * 16 + l16) * KSTG];
#pragma unroll
        for (int ks = 0; ks < 2; ++ks) {
            const int kk = ks * 32;
            const int k0 = s * KSTG + kk;
            v16bf a = a_frag(arow, kk, hi);
#pragma unroll
            for (int t = 0; t < 4; ++t) {
                const int nt = t * 16 + l16;
                v16bf bO = b_frag(&shO[buf][0], nt, KSTG, kk, hi);
                accO[t] = WMMA_BF16(accO[t], a, bO);
            }
            if (k0 < Dn) {   // Bv half only -> ct GEMM
#pragma unroll
                for (int t = 0; t < 4; ++t) {
                    const int n = n0 + t * 16 + l16;
                    v16bf bC = b_frag(Wctb16, n, Dn, k0, hi);
                    accC[t] = WMMA_BF16(accC[t], a, bC);
                }
            }
        }
        __syncthreads();
    }

#pragma unroll
    for (int t = 0; t < 4; ++t) {
        const int n = n0 + t * 16 + l16;
        const float bO = Wogb_b[n] + WogA_b[n];
        const float bC = Wctb_b[n];
#pragma unroll
        for (int r = 0; r < 8; ++r) {
            const int m = m0 + wave * 16 + r + (hi ? 8 : 0);
            og[(size_t)m * Dn + n]  = sigmoidf_(accO[t][r] + bO);
            ctb[(size_t)m * Dn + n] = tanhf(accC[t][r] + bC);
        }
    }
}

// ---------------------------------------------------------------------------
// Kernel 5: ct = fg*A + ig*ctb; ht = og*sigmoid(ct);
//           score = sigmoid(ht . final_w + final_b); out = A - (A-Bv)*score
// ---------------------------------------------------------------------------
__global__ void __launch_bounds__(256) finalize(
    const int* __restrict__ ip,
    const float* __restrict__ src,
    const float* __restrict__ A,
    const float* __restrict__ fg, const float* __restrict__ ig,
    const float* __restrict__ og, const float* __restrict__ ctb,
    const float* __restrict__ final_w, const float* __restrict__ final_b,
    float* __restrict__ out, float* __restrict__ score_out)
{
    const int i    = ip[0];
    const int wave = threadIdx.x >> 5;
    const int lane = threadIdx.x & 31;
    const int b    = blockIdx.x * 8 + wave;
    const size_t base = (size_t)b * Dn;
    const float* Bvr = src + ((size_t)b * Sn + i) * Dn;
    const int e0 = lane * 16;

    float av[16], bvv[16];
    float part = 0.f;
#pragma unroll
    for (int e = 0; e < 16; ++e) {
        const size_t idx = base + e0 + e;
        float a  = A[idx];       av[e]  = a;
        float bv = Bvr[e0 + e];  bvv[e] = bv;
        float ct = fg[idx] * a + ig[idx] * ctb[idx];
        float h  = og[idx] * sigmoidf_(ct);
        part += h * final_w[e0 + e];
    }
#pragma unroll
    for (int off = 16; off >= 1; off >>= 1)
        part += __shfl_xor(part, off, 32);
    const float score = sigmoidf_(part + final_b[0]);
#pragma unroll
    for (int e = 0; e < 16; ++e)
        out[base + e0 + e] = av[e] - (av[e] - bvv[e]) * score;
    if (lane == 0) score_out[b] = score;
}

// ---------------------------------------------------------------------------
extern "C" void kernel_launch(void* const* d_in, const int* in_sizes, int n_in,
                              void* d_out, int out_size, void* d_ws, size_t ws_size,
                              hipStream_t stream)
{
    (void)in_sizes; (void)n_in; (void)out_size; (void)ws_size;

    const int*   ip     = (const int*)  d_in[0];
    const float* src    = (const float*)d_in[1];
    const float* encm   = (const float*)d_in[2];
    const float* A      = (const float*)d_in[3];
    const float* sv     = (const float*)d_in[4];
    const float* Wfg_w  = (const float*)d_in[5];
    const float* Wfg_b  = (const float*)d_in[6];
    const float* Wig_w  = (const float*)d_in[7];
    const float* Wig_b  = (const float*)d_in[8];
    const float* Wogb_w = (const float*)d_in[9];
    const float* Wogb_b = (const float*)d_in[10];
    const float* WogA_w = (const float*)d_in[11];
    const float* WogA_b = (const float*)d_in[12];
    const float* Wctb_w = (const float*)d_in[13];
    const float* Wctb_b = (const float*)d_in[14];
    const float* fin_w  = (const float*)d_in[15];
    const float* fin_b  = (const float*)d_in[16];

    float* out       = (float*)d_out;
    float* score_out = out + (size_t)Bn * Dn;

    char* ws = (char*)d_ws;
    size_t off = 0;
    auto carve = [&](size_t bytes) -> void* {
        void* p = ws + off;
        off = (off + bytes + 255) & ~(size_t)255;
        return p;
    };
    float*  prev_cur = (float*) carve((size_t)Bn * 4);
    float*  prev_tot = (float*) carve((size_t)Bn * 4);
    bf16_t* enc16    = (bf16_t*)carve((size_t)Bn * K1 * 2);
    bf16_t* BA16     = (bf16_t*)carve((size_t)Bn * K1 * 2);
    bf16_t* Wfg16    = (bf16_t*)carve((size_t)Dn * K1 * 2);
    bf16_t* Wig16    = (bf16_t*)carve((size_t)Dn * K1 * 2);
    bf16_t* Wog16    = (bf16_t*)carve((size_t)Dn * K1 * 2);
    bf16_t* Wctb16   = (bf16_t*)carve((size_t)Dn * Dn * 2);
    float*  fg       = (float*) carve((size_t)Bn * Dn * 4);
    float*  ig       = (float*) carve((size_t)Bn * Dn * 4);
    float*  og       = (float*) carve((size_t)Bn * Dn * 4);
    float*  ctb      = (float*) carve((size_t)Bn * Dn * 4);

    prep_rows<<<Bn / 8, 256, 0, stream>>>(ip, src, encm, A, sv,
                                          enc16, BA16, prev_cur, prev_tot);

    conv_weights<<<(Dn * K1) / 256, 256, 0, stream>>>(
        Wfg_w, Wig_w, Wogb_w, WogA_w, Wctb_w, Wfg16, Wig16, Wog16, Wctb16);

    dim3 ggrid(Bn / 64, Dn / 64);
    gemm_fgig<<<ggrid, 128, 0, stream>>>(enc16, Wfg16, Wig16, Wfg_w, Wig_w,
                                         Wfg_b, Wig_b, prev_cur, prev_tot,
                                         fg, ig);
    gemm_ogct<<<ggrid, 128, 0, stream>>>(BA16, Wog16, Wctb16,
                                         Wogb_b, WogA_b, Wctb_b, og, ctb);

    finalize<<<Bn / 8, 256, 0, stream>>>(ip, src, A, fg, ig, og, ctb,
                                         fin_w, fin_b, out, score_out);
}
